// CrossAttention_67095979098756
// MI455X (gfx1250) — matrix operand order repaired and checked by reading, not compiled, not run on hardware
//
#include <hip/hip_runtime.h>
#include <hip/hip_bf16.h>

// ---------------------------------------------------------------------------
// Cross-attention for MI455X (gfx1250): bf16 WMMA + async global->LDS DMA.
//   out  = softmax((X Wq + bq)(Y Wk + bk)^T / sqrt(E), mask) (Y Wv + bv) Wo + bo
// d_out = [ out (B,QL,E) f32 | attn (B,QL,KL) f32 ]
// GEMM: workgroup tile 256x128, wave tile 64x64 (16 WMMA accumulators),
// triple-buffered async LDS staging, one barrier per K-stage.
// Workspace (~264 MiB).
// ---------------------------------------------------------------------------

typedef __attribute__((ext_vector_type(16))) __bf16       v16bf;
typedef __attribute__((ext_vector_type(8)))  float        v8f;
typedef __attribute__((ext_vector_type(8)))  unsigned int v8u;

union FragU { v8u u; v16bf b; };

__device__ __forceinline__ unsigned short f2bf(float f) {
    unsigned int u = __float_as_uint(f);
    u += 0x7FFFu + ((u >> 16) & 1u);          // round-to-nearest-even
    return (unsigned short)(u >> 16);
}

// Async DMA global -> LDS. IOFFSET is added to BOTH LDS and global addresses
// (ISA 08_async_tensor §4.4), so contiguous chunks need no extra address math.
__device__ __forceinline__ void async_copy32(unsigned ldsAddr, unsigned gOff,
                                             unsigned long long base) {
    asm volatile("global_load_async_to_lds_b128 %0, %1, %2\n\t"
                 "global_load_async_to_lds_b128 %0, %1, %2 offset:16"
                 :: "v"(ldsAddr), "v"(gOff), "s"(base) : "memory");
}
__device__ __forceinline__ void async_copy64(unsigned ldsAddr, unsigned gOff,
                                             unsigned long long base) {
    asm volatile("global_load_async_to_lds_b128 %0, %1, %2\n\t"
                 "global_load_async_to_lds_b128 %0, %1, %2 offset:16\n\t"
                 "global_load_async_to_lds_b128 %0, %1, %2 offset:32\n\t"
                 "global_load_async_to_lds_b128 %0, %1, %2 offset:48"
                 :: "v"(ldsAddr), "v"(gOff), "s"(base) : "memory");
}
__device__ __forceinline__ void wait_async_le6() {
    asm volatile("s_wait_asynccnt 0x6" ::: "memory");
}
__device__ __forceinline__ void wait_async_0() {
    asm volatile("s_wait_asynccnt 0x0" ::: "memory");
}

// ----------------------- f32 -> bf16 bulk convert ---------------------------
__global__ __launch_bounds__(256) void cvt_bf16x8(const float* __restrict__ src,
                                                  unsigned short* __restrict__ dst) {
    const size_t i = ((size_t)blockIdx.x * 256 + threadIdx.x) * 8;
    const float4* s4 = (const float4*)(src + i);
    float4 a = s4[0], b = s4[1];
    uint4 o;
    o.x = f2bf(a.x) | ((unsigned)f2bf(a.y) << 16);
    o.y = f2bf(a.z) | ((unsigned)f2bf(a.w) << 16);
    o.z = f2bf(b.x) | ((unsigned)f2bf(b.y) << 16);
    o.w = f2bf(b.z) | ((unsigned)f2bf(b.w) << 16);
    *(uint4*)(dst + i) = o;
}

// -------------------------- weight transpose (f32 -> bf16^T) ----------------
__global__ __launch_bounds__(256) void transpose_w(const float* __restrict__ src,
                                                   unsigned short* __restrict__ dst,
                                                   int n) {
    __shared__ float tile[32][33];
    const int bx = blockIdx.x * 32, by = blockIdx.y * 32;
    const int tx = threadIdx.x, ty = threadIdx.y;  // block (32, 8)
#pragma unroll
    for (int i = 0; i < 32; i += 8)
        tile[ty + i][tx] = src[(size_t)(by + ty + i) * n + bx + tx];
    __syncthreads();
#pragma unroll
    for (int i = 0; i < 32; i += 8)
        dst[(size_t)(bx + ty + i) * n + by + tx] = f2bf(tile[tx][ty + i]);
}

// ------------------------------- tiled WMMA GEMM ----------------------------
// C(M x N) = A(M x K) * B^T, A and Bt both bf16 row-major (Bt is N x K).
constexpr int EPI_PROJ_BF16 = 0;   // C bf16 = acc + bias          (row major)
constexpr int EPI_PROJ_VT   = 1;   // C bf16 = acc + bias          (transposed: Vt[b][n][m])
constexpr int EPI_SCORES    = 2;   // C f32  = acc * scale         (batched)
constexpr int EPI_ATT       = 3;   // C bf16 = acc                 (batched)
constexpr int EPI_OUT       = 4;   // C f32  = acc + bias

template <int EPI>
__global__ __launch_bounds__(256) void gemm256(const unsigned short* __restrict__ A,
                                               const unsigned short* __restrict__ Bt,
                                               const float* __restrict__ bias,
                                               void* __restrict__ Cout,
                                               int K, int ldc,
                                               long batchA, long batchB, long batchC,
                                               float scale, int rowsPerBatch) {
    __shared__ unsigned short lA[3][256 * 40];   // 32 bf16 data + 8 pad per row
    __shared__ unsigned short lB[3][128 * 40];

    const int tid   = threadIdx.x;
    const int lane  = tid & 31;
    const int wave  = tid >> 5;
    const int waveM = wave >> 1;        // 0..3 -> M offset waveM*64
    const int waveN = wave & 1;         // 0..1 -> N offset waveN*64
    const int half  = lane >> 4;        // 0..1
    const int r     = lane & 15;

    const long z = blockIdx.z;
    const unsigned long long Abase64 = (unsigned long long)(A + z * batchA);
    const unsigned long long Bbase64 = (unsigned long long)(Bt + z * batchB);

    const int blockM = blockIdx.y * 256;
    const int blockN = blockIdx.x * 128;

    // Staging: thread t copies full A row t (64B) and half of B row t>>1 (32B).
    const int aRow = tid;
    const int bRow = tid >> 1;
    const int bCol = (tid & 1) * 16;    // halfword offset

    unsigned aOff = (unsigned)(((size_t)(blockM + aRow) * K) * 2);
    unsigned bOff = (unsigned)(((size_t)(blockN + bRow) * K + bCol) * 2);

    const unsigned lA0 = (unsigned)(unsigned long long)&lA[0][aRow * 40];
    const unsigned lB0 = (unsigned)(unsigned long long)&lB[0][bRow * 40 + bCol];
    constexpr unsigned ASTRIDE = 256u * 40u * 2u;   // bytes per A buffer
    constexpr unsigned BSTRIDE = 128u * 40u * 2u;   // bytes per B buffer

    v8f zinit = {};
    v8f acc[4][4];
#pragma unroll
    for (int i = 0; i < 4; ++i)
#pragma unroll
        for (int j = 0; j < 4; ++j) acc[i][j] = zinit;

    auto compute = [&](int buf) {
        // B fragments: lane = column; low lanes K=0..15, high lanes K=16..31.
        FragU bfr[4];
#pragma unroll
        for (int nt = 0; nt < 4; ++nt) {
            const int row = waveN * 64 + nt * 16 + r;
            const uint4* p = (const uint4*)&lB[buf][row * 40];
            uint4 lo = p[half * 2];
            uint4 hi = p[half * 2 + 1];
            v8u v;
            v[0] = lo.x; v[1] = lo.y; v[2] = lo.z; v[3] = lo.w;
            v[4] = hi.x; v[5] = hi.y; v[6] = hi.z; v[7] = hi.w;
            bfr[nt].u = v;
        }
        // A fragments: lane = row; low lanes K=0..15, high lanes K=16..31.
#pragma unroll
        for (int mt = 0; mt < 4; ++mt) {
            const int row = waveM * 64 + mt * 16 + r;
            const uint4* p = (const uint4*)&lA[buf][row * 40];
            uint4 lo = p[half * 2];
            uint4 hi = p[half * 2 + 1];
            v8u v;
            v[0] = lo.x; v[1] = lo.y; v[2] = lo.z; v[3] = lo.w;
            v[4] = hi.x; v[5] = hi.y; v[6] = hi.z; v[7] = hi.w;
            FragU afr; afr.u = v;
#pragma unroll
            for (int nt = 0; nt < 4; ++nt) {
                acc[mt][nt] = __builtin_amdgcn_wmma_f32_16x16x32_bf16(
                    false, afr.b, false, bfr[nt].b, (short)0, acc[mt][nt],
                    false, false);
            }
        }
    };

    // Prologue: stage 0 -> buffer 0 (6 outstanding async DMAs / thread).
    async_copy64(lA0, aOff, Abase64);
    async_copy32(lB0, bOff, Bbase64);

    // Triple-buffered steady state, one barrier per stage:
    // issue(s+1 -> fill) | wait prev landed | barrier | compute(cur).
    // fill == buffer last read at stage s-2, protected by two barrier epochs.
    const int stages = K >> 5;
    int cur = 0;
    for (int s = 0; s < stages - 1; ++s) {
        int fill = cur + 1;
        if (fill == 3) fill = 0;
        async_copy64(lA0 + (unsigned)fill * ASTRIDE, aOff + 64, Abase64);
        async_copy32(lB0 + (unsigned)fill * BSTRIDE, bOff + 64, Bbase64);
        aOff += 64;
        bOff += 64;
        wait_async_le6();   // in-order completion => stage s fully landed
        __syncthreads();
        compute(cur);
        cur = fill;
    }
    wait_async_0();
    __syncthreads();
    compute(cur);

    // ------------------------------- epilogue -------------------------------
    long bbBase = 0, rrBase = 0;
    if constexpr (EPI == EPI_PROJ_VT) {
        bbBase = blockM / rowsPerBatch;          // 256-row block never straddles
        rrBase = blockM - bbBase * rowsPerBatch; // a batch boundary
    }
#pragma unroll
    for (int mt = 0; mt < 4; ++mt) {
#pragma unroll
        for (int nt = 0; nt < 4; ++nt) {
            const int col = blockN + waveN * 64 + nt * 16 + r;
            float bv_ = 0.0f;
            if constexpr (EPI == EPI_PROJ_BF16 || EPI == EPI_PROJ_VT || EPI == EPI_OUT)
                bv_ = bias[col];
#pragma unroll
            for (int v = 0; v < 8; ++v) {
                const int rowIn = waveM * 64 + mt * 16 + half * 8 + v;
                const int rowL = blockM + rowIn;
                const float val = acc[mt][nt][v];
                if constexpr (EPI == EPI_PROJ_BF16) {
                    ((unsigned short*)Cout)[(size_t)rowL * ldc + col] = f2bf(val + bv_);
                } else if constexpr (EPI == EPI_PROJ_VT) {
                    ((unsigned short*)Cout)[bbBase * batchC + (long)col * ldc +
                                            (rrBase + rowIn)] = f2bf(val + bv_);
                } else if constexpr (EPI == EPI_SCORES) {
                    ((float*)Cout)[z * batchC + (size_t)rowL * ldc + col] = val * scale;
                } else if constexpr (EPI == EPI_ATT) {
                    ((unsigned short*)Cout)[z * batchC + (size_t)rowL * ldc + col] =
                        f2bf(val);
                } else {  // EPI_OUT
                    ((float*)Cout)[(size_t)rowL * ldc + col] = val + bv_;
                }
            }
        }
    }
}

// ------------------------------ masked row softmax --------------------------
__global__ __launch_bounds__(256) void softmax_rows(float* __restrict__ scores,
                                                    const int* __restrict__ mask,
                                                    unsigned short* __restrict__ attnbf) {
    const int cols = 2048;
    const long row = blockIdx.x;
    float* s = scores + row * (long)cols;
    const int* m = mask + row * (long)cols;
    unsigned short* o = attnbf + row * (long)cols;
    const int tid = threadIdx.x;

    float v[8];
    float mx = -INFINITY;
#pragma unroll
    for (int i = 0; i < 8; ++i) {
        const int c = tid + i * 256;
        float x = s[c];
        x = m[c] ? x : -INFINITY;
        v[i] = x;
        mx = fmaxf(mx, x);
    }
#pragma unroll
    for (int off = 16; off > 0; off >>= 1) mx = fmaxf(mx, __shfl_xor(mx, off, 32));
    __shared__ float red[8];
    if ((tid & 31) == 0) red[tid >> 5] = mx;
    __syncthreads();
    float bm = red[0];
#pragma unroll
    for (int i = 1; i < 8; ++i) bm = fmaxf(bm, red[i]);
    __syncthreads();

    float e[8];
    float sum = 0.0f;
#pragma unroll
    for (int i = 0; i < 8; ++i) { e[i] = __expf(v[i] - bm); sum += e[i]; }
#pragma unroll
    for (int off = 16; off > 0; off >>= 1) sum += __shfl_xor(sum, off, 32);
    if ((tid & 31) == 0) red[tid >> 5] = sum;
    __syncthreads();
    float tot = 0.0f;
#pragma unroll
    for (int i = 0; i < 8; ++i) tot += red[i];
    const float inv = 1.0f / tot;
#pragma unroll
    for (int i = 0; i < 8; ++i) {
        const int c = tid + i * 256;
        const float p = e[i] * inv;
        s[c] = p;
        o[c] = f2bf(p);
    }
}

// --------------------------------- launcher ---------------------------------
extern "C" void kernel_launch(void* const* d_in, const int* in_sizes, int n_in,
                              void* d_out, int out_size, void* d_ws, size_t ws_size,
                              hipStream_t stream) {
    const float* query = (const float*)d_in[0];
    const float* keyv  = (const float*)d_in[1];
    const int*   mask  = (const int*)d_in[2];
    const float* Wq    = (const float*)d_in[3];
    const float* bq    = (const float*)d_in[4];
    const float* Wk    = (const float*)d_in[5];
    const float* bk    = (const float*)d_in[6];
    const float* Wv    = (const float*)d_in[7];
    const float* bv    = (const float*)d_in[8];
    const float* Wo    = (const float*)d_in[9];
    const float* bo    = (const float*)d_in[10];

    constexpr int B = 8, QL = 2048, KL = 2048, E = 1024, D = 1024;
    constexpr long ME = (long)B * QL;  // 16384 flat rows
    constexpr size_t NIN = (size_t)B * QL * D;  // 16,777,216 elements

    unsigned short* WqT  = (unsigned short*)d_ws;
    unsigned short* WkT  = WqT + (size_t)E * D;
    unsigned short* WvT  = WkT + (size_t)E * D;
    unsigned short* WoT  = WvT + (size_t)E * D;
    unsigned short* Qin  = WoT + (size_t)E * D;              // bf16 query
    unsigned short* KVin = Qin + NIN;                        // bf16 key_value
    unsigned short* Qbf  = KVin + NIN;
    unsigned short* Kbf  = Qbf + (size_t)B * QL * E;
    unsigned short* Vt   = Kbf + (size_t)B * KL * E;         // (B, E, KL)
    unsigned short* Abf  = Vt  + (size_t)B * E * KL;         // (B, QL, KL)
    unsigned short* Attd = Abf + (size_t)B * QL * KL;        // (B*QL, E)

    float* outp = (float*)d_out;
    float* attn = outp + (size_t)B * QL * E;

    // 0) inputs: f32 -> bf16
    cvt_bf16x8<<<dim3((unsigned)(NIN / 2048)), 256, 0, stream>>>(query, Qin);
    cvt_bf16x8<<<dim3((unsigned)(NIN / 2048)), 256, 0, stream>>>(keyv, KVin);

    // 1) weights: f32 (D x E) -> bf16 transposed (E x D)
    {
        dim3 tb(32, 8), tg(D / 32, D / 32);
        transpose_w<<<tg, tb, 0, stream>>>(Wq, WqT, D);
        transpose_w<<<tg, tb, 0, stream>>>(Wk, WkT, D);
        transpose_w<<<tg, tb, 0, stream>>>(Wv, WvT, D);
        transpose_w<<<tg, tb, 0, stream>>>(Wo, WoT, D);
    }

    // 2) projections
    gemm256<EPI_PROJ_BF16><<<dim3(E / 128, ME / 256, 1), 256, 0, stream>>>(
        Qin, WqT, bq, Qbf, D, E, 0, 0, 0, 1.0f, 0);
    gemm256<EPI_PROJ_BF16><<<dim3(E / 128, ME / 256, 1), 256, 0, stream>>>(
        KVin, WkT, bk, Kbf, D, E, 0, 0, 0, 1.0f, 0);
    gemm256<EPI_PROJ_VT><<<dim3(E / 128, ME / 256, 1), 256, 0, stream>>>(
        KVin, WvT, bv, Vt, D, KL, 0, 0, (long)E * KL, 1.0f, KL);

    // 3) scores = Q K^T * 1/sqrt(E)  -> f32 attn region (raw, pre-softmax)
    gemm256<EPI_SCORES><<<dim3(KL / 128, QL / 256, B), 256, 0, stream>>>(
        Qbf, Kbf, nullptr, attn, E, KL,
        (long)QL * E, (long)KL * E, (long)QL * KL, 0.03125f, 0);

    // 4) masked softmax: normalize in-place + bf16 copy
    softmax_rows<<<dim3((unsigned)ME), 256, 0, stream>>>(attn, mask, Abf);

    // 5) attended = attn @ V   (B-operand = Vt rows, contiguous)
    gemm256<EPI_ATT><<<dim3(E / 128, QL / 256, B), 256, 0, stream>>>(
        Abf, Vt, nullptr, Attd, KL, E,
        (long)QL * KL, (long)E * KL, (long)QL * E, 1.0f, 0);

    // 6) out = attended @ Wo + bo -> f32
    gemm256<EPI_OUT><<<dim3(E / 128, ME / 256, 1), 256, 0, stream>>>(
        Attd, WoT, bo, outp, E, E, 0, 0, 0, 1.0f, 0);
}
